// MultiHeadAttention_4844723110450
// MI455X (gfx1250) — matrix operand order repaired and checked by reading, not compiled, not run on hardware
//
#include <hip/hip_runtime.h>

typedef __attribute__((ext_vector_type(16))) __bf16 v16bf;
typedef __attribute__((ext_vector_type(8)))  __bf16 v8bf;
typedef __attribute__((ext_vector_type(4)))  __bf16 v4bf;
typedef __attribute__((ext_vector_type(8)))  float  v8f;
typedef __attribute__((ext_vector_type(4)))  float  v4f;

// ---- helpers -------------------------------------------------------------

__device__ __forceinline__ __bf16 f2bf(float f) {
    unsigned u = __builtin_bit_cast(unsigned, f);
    u += 0x7FFFu + ((u >> 16) & 1u);            // round-to-nearest-even
    unsigned short h = (unsigned short)(u >> 16);
    return __builtin_bit_cast(__bf16, h);
}

__device__ __forceinline__ float bf2f(__bf16 b) {
    unsigned short h = __builtin_bit_cast(unsigned short, b);
    unsigned u = ((unsigned)h) << 16;
    return __builtin_bit_cast(float, u);
}

__device__ __forceinline__ v16bf cat16(v8bf lo, v8bf hi) {
    return __builtin_shufflevector(lo, hi, 0,1,2,3,4,5,6,7,8,9,10,11,12,13,14,15);
}

__device__ __forceinline__ v8f zero8() {
    v8f z;
#pragma unroll
    for (int i = 0; i < 8; ++i) z[i] = 0.0f;
    return z;
}

__device__ __forceinline__ v8f wmma_bf16(v16bf a, v16bf b, v8f c) {
    return __builtin_amdgcn_wmma_f32_16x16x32_bf16(
        false, a, false, b, (short)0, c, false, false);
}

// gfx1250 async global->LDS copy (ASYNCcnt-tracked); inline asm: portable
// across ROCm-7.2 and amdgpu-toolchain (no builtin dependency).
__device__ __forceinline__ void async_ld_b128(unsigned lds_off, const void* gaddr) {
    asm volatile("global_load_async_to_lds_b128 %0, %1, off"
                 :: "v"(lds_off), "v"(gaddr) : "memory");
}
__device__ __forceinline__ void wait_async_le2() {
    asm volatile("s_wait_asynccnt 0x2" ::: "memory");
}
__device__ __forceinline__ void wait_async_0() {
    asm volatile("s_wait_asynccnt 0x0" ::: "memory");
}

// ---- kernel 0: elementwise f32 -> bf16 (vector x4) -----------------------
__global__ void convert_bf16(const float* __restrict__ src,
                             __bf16* __restrict__ dst, int n4) {
    int i = blockIdx.x * blockDim.x + threadIdx.x;
    if (i >= n4) return;
    v4f f = *(const v4f*)(src + (size_t)i * 4);
    v4bf o;
#pragma unroll
    for (int k = 0; k < 4; ++k) o[k] = f2bf(f[k]);
    *(v4bf*)(dst + (size_t)i * 4) = o;
}

// ---- kernel 1: weight transpose + f32->bf16 ------------------------------
__global__ void w_transpose_bf16(const float* __restrict__ src,
                                 __bf16* __restrict__ dst, int R, int C) {
    int idx = blockIdx.x * blockDim.x + threadIdx.x;
    if (idx >= R * C) return;
    int r = idx / C, c = idx % C;
    dst[(size_t)c * R + r] = f2bf(src[idx]);
}

// ---- kernel 2: QKV projection GEMM  Y = X[8192,256] @ W[256,512] ---------
// Xb: bf16 [8192,256], Wt = W^T bf16 [512,256].
// mode 0/1 -> q/k layout [bh][n][64]; mode 2 -> v transposed [bh][64][n].
__global__ __launch_bounds__(128) void proj_kernel(
    const __bf16* __restrict__ Xb, const __bf16* __restrict__ Wt,
    __bf16* __restrict__ Y, int mode) {
    const int lane = threadIdx.x & 31;
    const int wave = threadIdx.x >> 5;
    const int m16  = lane & 15;
    const int hi   = lane >> 4;
    const int M0   = blockIdx.x * 16;
    const int n0   = blockIdx.y * 256 + wave * 64;

    v8f acc[4];
#pragma unroll
    for (int t = 0; t < 4; ++t) acc[t] = zero8();

    const __bf16* xrow = Xb + (size_t)(M0 + m16) * 256;

    for (int kk = 0; kk < 256; kk += 32) {
        const __bf16* ar = xrow + kk + hi * 8;
        v16bf a = cat16(*(const v8bf*)ar, *(const v8bf*)(ar + 16));
#pragma unroll
        for (int t = 0; t < 4; ++t) {
            const __bf16* wr = Wt + (size_t)(n0 + t * 16 + m16) * 256 + kk + hi * 8;
            v16bf b = cat16(*(const v8bf*)wr, *(const v8bf*)(wr + 16));
            acc[t] = wmma_bf16(a, b, acc[t]);
        }
    }

#pragma unroll
    for (int t = 0; t < 4; ++t) {
#pragma unroll
        for (int r = 0; r < 8; ++r) {
            int m  = M0 + r + hi * 8;
            int nn = n0 + t * 16 + m16;
            int b = m >> 11, ntok = m & 2047;
            int h = nn >> 6, u = nn & 63;
            int bh = b * 8 + h;
            size_t idx = (mode == 2)
                ? ((size_t)(bh * 64 + u) * 2048 + ntok)
                : ((size_t)(bh * 2048 + ntok) * 64 + u);
            Y[idx] = f2bf(acc[t][r]);
        }
    }
}

// ---- kernel 3: flash attention, async-LDS staged K/V ---------------------
// q,k: [32][2048][64] bf16   v: [32][64][2048] bf16
// head out: [8192][512] bf16  (= softmax(qk^T/8) v + q)
// Block = 8 waves, one (b,h); each wave owns 32 q-rows; K/V blocks staged
// once per block into LDS with global_load_async_to_lds_b128 (double-buffer).
__global__ __launch_bounds__(256) void attn_kernel(
    const __bf16* __restrict__ q16, const __bf16* __restrict__ k16,
    const __bf16* __restrict__ v16t, __bf16* __restrict__ head) {
    __shared__ __bf16 sK[2][32 * 64];          // 2 x 4KB: K rows j..j+31
    __shared__ __bf16 sV[2][64 * 32];          // 2 x 4KB: V^T rows u, cols j..
    __shared__ __bf16 sP[8][32 * 32];          // 8 x 2KB: per-wave P tile
    const int tid   = threadIdx.x;
    const int lane  = tid & 31;
    const int wave  = tid >> 5;
    const int m16   = lane & 15;
    const int hi    = lane >> 4;
    const int bh    = blockIdx.y;
    const int qbase = blockIdx.x * 256 + wave * 32;

    const __bf16* qp = q16 + (size_t)bh * 2048 * 64;
    const __bf16* kp = k16 + (size_t)bh * 2048 * 64;
    const __bf16* vp = v16t + (size_t)bh * 64 * 2048;
    __bf16* ldsW = sP[wave];

    // per-thread staging chunks (16B each): K 32x64 = 256 chunks, V 64x32 too
    const int krow = tid >> 3, kc = (tid & 7) * 8;
    const int vrow = tid >> 2, vc = (tid & 3) * 8;

    // q A-fragments: 2 row-tiles x 2 K-chunks
    v16bf qa[2][2];
#pragma unroll
    for (int rt = 0; rt < 2; ++rt)
#pragma unroll
        for (int c = 0; c < 2; ++c) {
            const __bf16* qr =
                qp + (size_t)(qbase + rt * 16 + m16) * 64 + c * 32 + hi * 8;
            qa[rt][c] = cat16(*(const v8bf*)qr, *(const v8bf*)(qr + 16));
        }

    v8f o[2][4];
#pragma unroll
    for (int rt = 0; rt < 2; ++rt)
#pragma unroll
        for (int t = 0; t < 4; ++t) o[rt][t] = zero8();
    float mrow[16], lrow[16];
#pragma unroll
    for (int r = 0; r < 16; ++r) { mrow[r] = -__builtin_inff(); lrow[r] = 0.0f; }

    constexpr float SCALE = 0.125f;            // 1/sqrt(64)
    constexpr float L2E = 1.44269504088896340736f;

    // prologue: stage block j=0 into buffer 0
    async_ld_b128((unsigned)(size_t)(&sK[0][krow * 64 + kc]),
                  kp + (size_t)krow * 64 + kc);
    async_ld_b128((unsigned)(size_t)(&sV[0][vrow * 32 + vc]),
                  vp + (size_t)vrow * 2048 + vc);

    for (int j = 0; j < 2048; j += 32) {
        const int cur = (j >> 5) & 1, nxt = cur ^ 1;
        if (j + 32 < 2048) {                   // stage next block
            async_ld_b128((unsigned)(size_t)(&sK[nxt][krow * 64 + kc]),
                          kp + (size_t)(j + 32 + krow) * 64 + kc);
            async_ld_b128((unsigned)(size_t)(&sV[nxt][vrow * 32 + vc]),
                          vp + (size_t)vrow * 2048 + (j + 32) + vc);
            wait_async_le2();                  // in-order: current 2 landed
        } else {
            wait_async_0();
        }
        __syncthreads();                       // all waves' staging visible

        // K^T B-fragments from LDS: 2 col-tiles x 2 K-chunks
        v16bf kb[2][2];
#pragma unroll
        for (int ct = 0; ct < 2; ++ct)
#pragma unroll
            for (int c = 0; c < 2; ++c) {
                const __bf16* kr =
                    &sK[cur][(ct * 16 + m16) * 64 + c * 32 + hi * 8];
                kb[ct][c] = cat16(*(const v8bf*)kr, *(const v8bf*)(kr + 16));
            }
        // scores: 2x2 tiles of 16x16
        v8f s[2][2];
#pragma unroll
        for (int rt = 0; rt < 2; ++rt)
#pragma unroll
            for (int ct = 0; ct < 2; ++ct) {
                v8f a = zero8();
#pragma unroll
                for (int c = 0; c < 2; ++c)
                    a = wmma_bf16(qa[rt][c], kb[ct][c], a);
                s[rt][ct] = a;
            }

        // online softmax per row (row = rt*16 + r + hi*8, col = m16)
#pragma unroll
        for (int rt = 0; rt < 2; ++rt) {
            float corr[8];
#pragma unroll
            for (int r = 0; r < 8; ++r) {
                int ri = rt * 8 + r;
                float a0 = s[rt][0][r] * SCALE, a1 = s[rt][1][r] * SCALE;
                float t = fmaxf(a0, a1);
#pragma unroll
                for (int msk = 1; msk < 16; msk <<= 1)
                    t = fmaxf(t, __shfl_xor(t, msk, 32));
                float mnew = fmaxf(mrow[ri], t);
                float p0 = exp2f((a0 - mnew) * L2E);
                float p1 = exp2f((a1 - mnew) * L2E);
                float c  = exp2f((mrow[ri] - mnew) * L2E);
                float ps = p0 + p1;
#pragma unroll
                for (int msk = 1; msk < 16; msk <<= 1)
                    ps += __shfl_xor(ps, msk, 32);
                lrow[ri] = lrow[ri] * c + ps;
                mrow[ri] = mnew;
                corr[r]  = c;
                s[rt][0][r] = p0;
                s[rt][1][r] = p1;
            }
#pragma unroll
            for (int t = 0; t < 4; ++t)
#pragma unroll
                for (int r = 0; r < 8; ++r) o[rt][t][r] *= corr[r];
            // C-layout P -> LDS (wave-private slice)
#pragma unroll
            for (int r = 0; r < 8; ++r) {
                ldsW[(rt * 16 + r + hi * 8) * 32 + m16]      = f2bf(s[rt][0][r]);
                ldsW[(rt * 16 + r + hi * 8) * 32 + 16 + m16] = f2bf(s[rt][1][r]);
            }
        }
        __builtin_amdgcn_wave_barrier();       // DS ops in-order per wave
        v16bf pa[2];
#pragma unroll
        for (int rt = 0; rt < 2; ++rt) {
            const __bf16* pr = ldsW + (rt * 16 + m16) * 32 + hi * 8;
            pa[rt] = cat16(*(const v8bf*)pr, *(const v8bf*)(pr + 16));
        }
        __builtin_amdgcn_wave_barrier();

        // o += P @ V[j..j+31, :]  (V^T from LDS: contiguous K, B reused x2)
#pragma unroll
        for (int t = 0; t < 4; ++t) {
            const __bf16* vr = &sV[cur][(t * 16 + m16) * 32 + hi * 8];
            v16bf b = cat16(*(const v8bf*)vr, *(const v8bf*)(vr + 16));
#pragma unroll
            for (int rt = 0; rt < 2; ++rt)
                o[rt][t] = wmma_bf16(pa[rt], b, o[rt][t]);
        }
        __syncthreads();                       // buffer reusable next+1 iter
    }

    // epilogue: normalize, +q residual, store [8192,512] bf16
    const int b = bh >> 3, h = bh & 7;
#pragma unroll
    for (int rt = 0; rt < 2; ++rt)
#pragma unroll
        for (int r = 0; r < 8; ++r) {
            float inv = 1.0f / lrow[rt * 8 + r];
            int row = qbase + rt * 16 + r + hi * 8;
#pragma unroll
            for (int t = 0; t < 4; ++t) {
                int u = t * 16 + m16;
                float val = o[rt][t][r] * inv + bf2f(qp[(size_t)row * 64 + u]);
                head[(size_t)(b * 2048 + row) * 512 + h * 64 + u] = f2bf(val);
            }
        }
}

// ---- kernel 4: output projection  out = head[8192,512] @ Wo[512,256] + bo
__global__ __launch_bounds__(128) void outproj_kernel(
    const __bf16* __restrict__ head, const __bf16* __restrict__ WoT,
    const float* __restrict__ bo, float* __restrict__ out) {
    const int lane = threadIdx.x & 31;
    const int wave = threadIdx.x >> 5;
    const int m16  = lane & 15;
    const int hi   = lane >> 4;
    const int M0   = blockIdx.x * 16;
    const int n0   = wave * 64;

    v8f acc[4];
#pragma unroll
    for (int t = 0; t < 4; ++t) acc[t] = zero8();

    const __bf16* arow = head + (size_t)(M0 + m16) * 512;
    for (int kk = 0; kk < 512; kk += 32) {
        const __bf16* ar = arow + kk + hi * 8;
        v16bf a = cat16(*(const v8bf*)ar, *(const v8bf*)(ar + 16));
#pragma unroll
        for (int t = 0; t < 4; ++t) {
            const __bf16* wr = WoT + (size_t)(n0 + t * 16 + m16) * 512 + kk + hi * 8;
            v16bf b = cat16(*(const v8bf*)wr, *(const v8bf*)(wr + 16));
            acc[t] = wmma_bf16(a, b, acc[t]);
        }
    }
#pragma unroll
    for (int t = 0; t < 4; ++t) {
#pragma unroll
        for (int r = 0; r < 8; ++r) {
            int c = n0 + t * 16 + m16;
            out[(size_t)(M0 + r + hi * 8) * 256 + c] = acc[t][r] + bo[c];
        }
    }
}

// ---- launch --------------------------------------------------------------

extern "C" void kernel_launch(void* const* d_in, const int* in_sizes, int n_in,
                              void* d_out, int out_size, void* d_ws, size_t ws_size,
                              hipStream_t stream) {
    (void)in_sizes; (void)n_in; (void)out_size; (void)ws_size;
    const float* Xq = (const float*)d_in[0];
    const float* Xk = (const float*)d_in[1];
    const float* Xv = (const float*)d_in[2];
    const float* Wq = (const float*)d_in[3];
    const float* Wk = (const float*)d_in[4];
    const float* Wv = (const float*)d_in[5];
    const float* Wo = (const float*)d_in[6];
    const float* bo = (const float*)d_in[7];
    float* out = (float*)d_out;

    char* ws = (char*)d_ws;
    const size_t SZ_X    = (size_t)8192 * 256 * 2;       // 4 MB each
    const size_t SZ_QKV  = (size_t)32 * 2048 * 64 * 2;   // 8 MB each
    const size_t SZ_HEAD = (size_t)8192 * 512 * 2;       // 8 MB
    const size_t SZ_W    = (size_t)512 * 256 * 2;        // 256 KB
    __bf16* q16  = (__bf16*)(ws);                 ws += SZ_QKV;
    __bf16* k16  = (__bf16*)(ws);                 ws += SZ_QKV;
    __bf16* v16t = (__bf16*)(ws);                 ws += SZ_QKV;
    __bf16* head = (__bf16*)(ws);                 ws += SZ_HEAD;
    __bf16* Xqb  = (__bf16*)(ws);                 ws += SZ_X;
    __bf16* Xkb  = (__bf16*)(ws);                 ws += SZ_X;
    __bf16* Xvb  = (__bf16*)(ws);                 ws += SZ_X;
    __bf16* WtQ  = (__bf16*)(ws);                 ws += SZ_W;
    __bf16* WtK  = (__bf16*)(ws);                 ws += SZ_W;
    __bf16* WtV  = (__bf16*)(ws);                 ws += SZ_W;
    __bf16* WoT  = (__bf16*)(ws);                 ws += SZ_W;

    const int XN4 = 8192 * 256 / 4;
    convert_bf16<<<(XN4 + 255) / 256, 256, 0, stream>>>(Xq, Xqb, XN4);
    convert_bf16<<<(XN4 + 255) / 256, 256, 0, stream>>>(Xk, Xkb, XN4);
    convert_bf16<<<(XN4 + 255) / 256, 256, 0, stream>>>(Xv, Xvb, XN4);

    const int WCNT = 256 * 512;
    w_transpose_bf16<<<(WCNT + 255) / 256, 256, 0, stream>>>(Wq, WtQ, 256, 512);
    w_transpose_bf16<<<(WCNT + 255) / 256, 256, 0, stream>>>(Wk, WtK, 256, 512);
    w_transpose_bf16<<<(WCNT + 255) / 256, 256, 0, stream>>>(Wv, WtV, 256, 512);
    w_transpose_bf16<<<(WCNT + 255) / 256, 256, 0, stream>>>(Wo, WoT, 512, 256);

    dim3 pgrid(512, 2, 1);
    proj_kernel<<<pgrid, 128, 0, stream>>>(Xqb, WtQ, q16, 0);
    proj_kernel<<<pgrid, 128, 0, stream>>>(Xkb, WtK, k16, 1);
    proj_kernel<<<pgrid, 128, 0, stream>>>(Xvb, WtV, v16t, 2);

    attn_kernel<<<dim3(8, 32), 256, 0, stream>>>(q16, k16, v16t, head);

    outproj_kernel<<<512, 128, 0, stream>>>(head, WoT, bo, out);
}
